// WordRNN_25297357373664
// MI455X (gfx1250) — matrix operand order repaired and checked by reading, not compiled, not run on hardware
//
#include <hip/hip_runtime.h>

// ---------------- problem constants ----------------
#define VOCAB 100000
#define EMB   20
#define HID   10
#define SEQ   100
#define NLAB  15
#define BATCH 16384
#define G4    (4*HID)          // 40 gate columns
#define NT    3                // 3 N-tiles of 16 (covers 40, last 8 padded)

#define WAVES_PER_BLOCK 8
#define ROWS_PER_WAVE   16
#define ASTRIDE 17             // 16 u32 A-pairs per row, padded (odd -> no bank conflicts)
#define ZSTRIDE 49             // 48 z-cols per row, padded (odd -> no bank conflicts)

#define LOG2E  1.4426950408889634f

typedef __attribute__((ext_vector_type(16))) _Float16 v16h;
typedef __attribute__((ext_vector_type(8)))  float    v8f;

__device__ __forceinline__ unsigned int pkf16(float a, float b) {
    union { _Float16 h[2]; unsigned int u; } cv;
    cv.h[0] = (_Float16)a;     // even K in [15:0]
    cv.h[1] = (_Float16)b;     // odd  K in [31:16]
    return cv.u;
}

__device__ __forceinline__ unsigned short f16bits(float a) {
    union { _Float16 h; unsigned short u; } cv;
    cv.h = (_Float16)a;
    return cv.u;
}

// Branch-free activations: one v_exp_f32 + one v_rcp_f32 each.
// Exact at extremes: exp2->inf => rcp->0 ; exp2->0 => plain limit. No clamps needed.
__device__ __forceinline__ float fast_sigmoid(float x) {
    return __builtin_amdgcn_rcpf(1.0f + __builtin_amdgcn_exp2f(-LOG2E * x));
}
__device__ __forceinline__ float fast_tanh(float x) {
    return 1.0f - 2.0f * __builtin_amdgcn_rcpf(1.0f + __builtin_amdgcn_exp2f((2.0f * LOG2E) * x));
}

// All LDS traffic here is cross-lane within ONE wave32 (tiles are wave-private).
// ISA guarantees LDS ops from a wave complete IN ORDER (DScnt, §7.3), so no
// hardware barrier is needed -- only a compiler scheduling fence so DS ops are
// not reordered across the exchange point. Zero instructions emitted, and no
// __syncthreads() acquire semantics draining loadcnt (which was serializing the
// prefetched embedding gather back into the critical path).
__device__ __forceinline__ void wave_lds_sync() {
    __builtin_amdgcn_wave_barrier();
    asm volatile("" ::: "memory");
}

__global__ __launch_bounds__(32 * WAVES_PER_BLOCK)
void wordrnn_lstm_wmma_kernel(const int*   __restrict__ x,      // [BATCH, SEQ]
                              const float* __restrict__ emb,    // [VOCAB, EMB]
                              const float* __restrict__ Wx,     // [EMB, 4H]
                              const float* __restrict__ Wh,     // [HID, 4H]
                              const float* __restrict__ bias,   // [4H]
                              const float* __restrict__ Wd,     // [HID, NLAB]
                              const float* __restrict__ bd,     // [NLAB]
                              float*       __restrict__ out)    // [BATCH, NLAB]
{
    __shared__ unsigned int abuf[WAVES_PER_BLOCK][ROWS_PER_WAVE][ASTRIDE];
    __shared__ float        zbuf[WAVES_PER_BLOCK][ROWS_PER_WAVE][ZSTRIDE];

    const int lane = threadIdx.x & 31;
    const int wave = threadIdx.x >> 5;
    const int m    = lane & 15;        // row within tile / column within N-tile
    const int hi   = lane >> 4;        // lane half (K sub-range; also hidden-unit half)
    const int tileRow0 = (blockIdx.x * WAVES_PER_BLOCK + wave) * ROWS_PER_WAVE;
    const int row  = tileRow0 + m;

    // ---- constant B fragments: Wfull = [Wx; Wh; 0; 0]  (32 x 48, cols >= 40 zero) ----
    // B 32x16 f16, contiguous k order used here (same as A): lanes 0-15 hold K=0..15 (2/VGPR), lanes 16-31 hold K=16..31; N = lane&15.
    v16h bfrag[NT];
    float biasv[NT];
    {
        const int kb = hi * 16;
        #pragma unroll
        for (int t = 0; t < NT; ++t) {
            const int g = t * 16 + m;
            #pragma unroll
            for (int v = 0; v < 8; ++v) {
                float w0 = 0.f, w1 = 0.f;
                if (g < G4) {
                    const int k0 = kb + 2 * v, k1 = k0 + 1;
                    w0 = (k0 < EMB) ? Wx[k0 * G4 + g]
                       : (k0 < EMB + HID) ? Wh[(k0 - EMB) * G4 + g] : 0.f;
                    w1 = (k1 < EMB) ? Wx[k1 * G4 + g]
                       : (k1 < EMB + HID) ? Wh[(k1 - EMB) * G4 + g] : 0.f;
                }
                bfrag[t][2 * v]     = (_Float16)w0;
                bfrag[t][2 * v + 1] = (_Float16)w1;
            }
            biasv[t] = (g < G4) ? bias[g] : 0.f;
        }
    }

    // Hidden state split across lane halves: lane m owns units 0..4 of its row,
    // lane m+16 owns units 5..9. Both halves run the full gate chain in parallel.
    float hs[5], cs[5];
    #pragma unroll
    for (int j = 0; j < 5; ++j) { hs[j] = 0.f; cs[j] = 0.f; }

    unsigned int* const arow = &abuf[wave][m][0];
    const int* const xrow = x + row * SEQ;

    // ---- two-deep software pipeline for the embedding gather ----
    // idx is prefetched TWO steps ahead, the embedding row ONE step ahead, so the
    // dependent idx->gather chain never produces an in-step s_wait: each load has
    // a full recurrence step (3 WMMA + ~50 TRANS ops) of slack before its use.
    float en[10];
    int   idxn;
    {
        const int idx0 = xrow[0];
        const float2* p2 = (const float2*)(emb + (size_t)idx0 * EMB + hi * 10);
        #pragma unroll
        for (int q = 0; q < 5; ++q) { float2 v2 = p2[q]; en[2*q] = v2.x; en[2*q+1] = v2.y; }
        idxn = xrow[1 < SEQ ? 1 : 0];                   // index for t=1
    }

    for (int t = 0; t < SEQ; ++t) {
        // ---- 1. pack A row vector (K=0..31): [e(20) | h(10) | 0 0] as f16 pairs ----
        #pragma unroll
        for (int q = 0; q < 5; ++q)
            arow[hi * 5 + q] = pkf16(en[2*q], en[2*q+1]);       // pairs 0..4 / 5..9
        // h pairs 10..14 (K=20..29); pair 12 straddles the unit split (h4|h5).
        unsigned short* const arow16 = (unsigned short*)arow;
        if (hi == 0) {
            arow[10] = pkf16(hs[0], hs[1]);
            arow[11] = pkf16(hs[2], hs[3]);
            arow16[2 * 12 + 0] = f16bits(hs[4]);                // K=24 (low half of pair 12)
        } else {
            arow16[2 * 12 + 1] = f16bits(hs[0]);                // K=25 (high half of pair 12)
            arow[13] = pkf16(hs[1], hs[2]);
            arow[14] = pkf16(hs[3], hs[4]);
            arow[15] = 0u;                                      // K=30,31 zero pad
        }

        // ---- 1b. issue next gather (idx already resident) + idx load for t+2 ----
        {
            const float2* p2 = (const float2*)(emb + (size_t)idxn * EMB + hi * 10);
            #pragma unroll
            for (int q = 0; q < 5; ++q) { float2 v2 = p2[q]; en[2*q] = v2.x; en[2*q+1] = v2.y; }
            const int tn2 = (t + 2 < SEQ) ? (t + 2) : (SEQ - 1);
            idxn = xrow[tn2];
        }
        wave_lds_sync();

        // ---- 2. read A fragment (contiguous k = 16*hi + s, same order as bfrag) ----
        // lane half 0: pairs {0..7}; half 1: pairs {8..15}
        v16h a;
        #pragma unroll
        for (int v = 0; v < 8; ++v) {
            const int pi = v + hi * 8;
            union { unsigned int u; _Float16 h2[2]; } cv;
            cv.u = arow[pi];
            a[2 * v]     = cv.h2[0];
            a[2 * v + 1] = cv.h2[1];
        }

        // ---- 3. z = [e|h] @ [Wx;Wh] + b : one fused WMMA per 16-col tile ----
        v8f z0, z1, z2;
        #pragma unroll
        for (int r = 0; r < 8; ++r) { z0[r] = biasv[0]; z1[r] = biasv[1]; z2[r] = biasv[2]; }
        z0 = __builtin_amdgcn_wmma_f32_16x16x32_f16(false, a, false, bfrag[0], (short)0, z0, false, false);
        z1 = __builtin_amdgcn_wmma_f32_16x16x32_f16(false, a, false, bfrag[1], (short)0, z1, false, false);
        z2 = __builtin_amdgcn_wmma_f32_16x16x32_f16(false, a, false, bfrag[2], (short)0, z2, false, false);

        // ---- 4. scatter C tiles to LDS: C layout -> row = hi*8 + r, col = tile*16 + (lane&15)
        #pragma unroll
        for (int r = 0; r < 8; ++r) {
            float* rowp = &zbuf[wave][hi * 8 + r][0];
            rowp[ 0 + m] = z0[r];
            rowp[16 + m] = z1[r];
            rowp[32 + m] = z2[r];
        }
        wave_lds_sync();

        // ---- 5. gates (Keras order i,f,c,o), branch-free TRANS math, both halves active
        {
            const float* zr = &zbuf[wave][m][0];
            const int jb = hi * 5;                    // this lane's unit base
            #pragma unroll
            for (int j = 0; j < 5; ++j) {
                const int u  = jb + j;
                const float ig = fast_sigmoid(zr[u]);
                const float fg = fast_sigmoid(zr[HID + u]);
                const float og = fast_sigmoid(zr[3 * HID + u]);
                cs[j] = fg * cs[j] + ig * fast_tanh(zr[2 * HID + u]);
                hs[j] = og * fast_tanh(cs[j]);
            }
        }
        wave_lds_sync();   // keep next step's abuf stores ordered after these reads
    }

    // ---- dense head: logits = h @ Wd + bd, split-K partials combined via LDS ----
    float part[NLAB];
    #pragma unroll
    for (int j = 0; j < NLAB; ++j) {
        float acc = 0.f;
        #pragma unroll
        for (int k = 0; k < 5; ++k)
            acc += hs[k] * Wd[(hi * 5 + k) * NLAB + j];
        part[j] = acc;
    }
    wave_lds_sync();
    if (hi == 1) {
        #pragma unroll
        for (int j = 0; j < NLAB; ++j) zbuf[wave][m][j] = part[j];
    }
    wave_lds_sync();
    if (hi == 0 && row < BATCH) {
        #pragma unroll
        for (int j = 0; j < NLAB; ++j)
            out[row * NLAB + j] = bd[j] + part[j] + zbuf[wave][m][j];
    }
}

extern "C" void kernel_launch(void* const* d_in, const int* in_sizes, int n_in,
                              void* d_out, int out_size, void* d_ws, size_t ws_size,
                              hipStream_t stream) {
    (void)in_sizes; (void)n_in; (void)d_ws; (void)ws_size; (void)out_size;
    const int*   x   = (const int*)  d_in[0];
    const float* emb = (const float*)d_in[1];
    const float* Wx  = (const float*)d_in[2];
    const float* Wh  = (const float*)d_in[3];
    const float* b   = (const float*)d_in[4];
    const float* Wd  = (const float*)d_in[5];
    const float* bd  = (const float*)d_in[6];
    // d_in[7] = drop_rate == 0 -> identity, ignored.
    float* out = (float*)d_out;

    const int rows_per_block = WAVES_PER_BLOCK * ROWS_PER_WAVE;   // 128
    const int grid = BATCH / rows_per_block;                      // 128 blocks
    wordrnn_lstm_wmma_kernel<<<grid, 32 * WAVES_PER_BLOCK, 0, stream>>>(
        x, emb, Wx, Wh, b, Wd, bd, out);
}
